// SelfAttentionModule_42923903156901
// MI455X (gfx1250) — matrix operand order repaired and checked by reading, not compiled, not run on hardware
//
#include <hip/hip_runtime.h>
#include <hip/hip_bf16.h>
#include <stdint.h>

#define BATCH 4
#define NPIX  4096
#define CDIM  256
#define CRDIM 32

typedef __bf16 bf16_t;
typedef bf16_t  v2bf  __attribute__((ext_vector_type(2)));
typedef bf16_t  v8bf  __attribute__((ext_vector_type(8)));
typedef bf16_t  v16bf __attribute__((ext_vector_type(16)));
typedef float   v8f   __attribute__((ext_vector_type(8)));
typedef uint32_t v8u  __attribute__((ext_vector_type(8)));

static __device__ __forceinline__ v16bf cat8(v8bf lo, v8bf hi) {
  return __builtin_shufflevector(lo, hi, 0,1,2,3,4,5,6,7,8,9,10,11,12,13,14,15);
}
static __device__ __forceinline__ v8f wmma_bf16(v16bf a, v16bf b, v8f c) {
  return __builtin_amdgcn_wmma_f32_16x16x32_bf16(false, a, false, b, (short)0, c,
                                                 false, false);
}
static __device__ __forceinline__ uint32_t pack2(float a, float b) {
  v2bf p; p[0] = (bf16_t)a; p[1] = (bf16_t)b;
  return __builtin_bit_cast(uint32_t, p);
}

// q is pre-scaled by (1/sqrt(32))*log2(e): scores come out in log2 domain,
// so softmax uses v_exp_f32 (exp2) directly with clip bounds +-10*log2(e).
#define QSCALE (0.17677669529663687f * 1.4426950408889634f)
#define CLIP2  14.426950408889634f

// ---------------------------------------------------------------------------
// Weight conversion f32 -> bf16 (row-major [o][c], c contiguous = A-operand).
// ---------------------------------------------------------------------------
__global__ __launch_bounds__(256) void wconv_kernel(
    const float* __restrict__ qw, const float* __restrict__ kw,
    const float* __restrict__ vw, const float* __restrict__ ow,
    bf16_t* __restrict__ wq, bf16_t* __restrict__ wk,
    bf16_t* __restrict__ wv, bf16_t* __restrict__ wo)
{
  const int i = blockIdx.x * 256 + threadIdx.x;     // grid 256 -> 65536 threads
  if (i < CRDIM * CDIM) { wq[i] = (bf16_t)qw[i]; wk[i] = (bf16_t)kw[i]; }
  wv[i] = (bf16_t)vw[i];
  wo[i] = (bf16_t)ow[i];
}

// ---------------------------------------------------------------------------
// x [b][c][n] f32  ->  xT [(b*N+n)][c] bf16 via LDS 64x64 tile transpose.
// ---------------------------------------------------------------------------
__global__ __launch_bounds__(256) void xpose_kernel(
    const float* __restrict__ x, bf16_t* __restrict__ xT)
{
  __shared__ bf16_t tile[64][65];
  const int blk = blockIdx.x;                       // BATCH * 4 * 64
  const int ng = blk & 63;
  const int cg = (blk >> 6) & 3;
  const int b  = blk >> 8;
  const int n0 = ng * 64, c0 = cg * 64;
  const int tid = threadIdx.x;
  const int nl = tid & 63, cl = tid >> 6;

  #pragma unroll
  for (int r = 0; r < 16; ++r) {
    int c = r * 4 + cl;
    tile[c][nl] = (bf16_t)x[(size_t)(b * CDIM + c0 + c) * NPIX + n0 + nl];
  }
  __syncthreads();

  const int n  = tid >> 2;
  const int ch = (tid & 3) * 16;
  v16bf v;
  #pragma unroll
  for (int e = 0; e < 16; ++e) v[e] = tile[ch + e][n];
  *(v16bf*)(xT + (size_t)(b * NPIX + n0 + n) * CDIM + c0 + ch) = v;
}

// ---------------------------------------------------------------------------
// WMMA q/k/v projections.  One block = one 16-pixel n-tile, 4 waves covering
// 20 o-tiles (q:2, k:2, v:16).  A = weight rows (M=o, K=c), B = xT rows
// (K=c, N=n), K=256 -> 8 WMMAs per tile.  The o-tile loop is software-
// pipelined: next tile's 8 A-tiles are loaded while current tile's WMMAs run.
// ---------------------------------------------------------------------------
__global__ __launch_bounds__(128) void proj_kernel(
    const bf16_t* __restrict__ xT,
    const bf16_t* __restrict__ wq, const bf16_t* __restrict__ wk,
    const bf16_t* __restrict__ wv,
    const float* __restrict__ qb, const float* __restrict__ kb,
    const float* __restrict__ vb,
    bf16_t* __restrict__ qT, bf16_t* __restrict__ kT, bf16_t* __restrict__ vbf)
{
  const int tile = blockIdx.x;                      // BATCH * 256
  const int b    = tile >> 8;
  const int n0   = (tile & 255) << 4;
  const int lane = threadIdx.x & 31;
  const int wave = threadIdx.x >> 5;
  const int half = lane >> 4;
  const int l16  = lane & 15;

  // Preload the 8 B-operand tiles of this n-tile (x row, c contiguous).
  const bf16_t* xrow = xT + (size_t)(b * NPIX + n0 + l16) * CDIM + (half ? 16 : 0);
  v16bf bx[8];
  #pragma unroll
  for (int kc = 0; kc < 8; ++kc) bx[kc] = *(const v16bf*)(xrow + kc * 32);

  auto sel = [&](int t, const bf16_t*& W, const float*& bias, int& o0, int& mode) {
    if (t < 2)      { W = wq; bias = qb; o0 = t * 16;       mode = 0; }
    else if (t < 4) { W = wk; bias = kb; o0 = (t - 2) * 16; mode = 1; }
    else            { W = wv; bias = vb; o0 = (t - 4) * 16; mode = 2; }
  };
  auto load_aw = [&](const bf16_t* W, int o0, v16bf* aw) {
    const bf16_t* wr = W + (size_t)(o0 + l16) * CDIM + (half ? 16 : 0);
    #pragma unroll
    for (int kc = 0; kc < 8; ++kc)
      aw[kc] = *(const v16bf*)(wr + kc * 32);
  };

  const bf16_t* W; const float* bias; int o0, mode;
  sel(wave * 5, W, bias, o0, mode);
  v16bf awc[8];
  load_aw(W, o0, awc);

  #pragma unroll
  for (int s = 0; s < 5; ++s) {
    const bf16_t* Wn = nullptr; const float* biasn = nullptr; int o0n = 0, moden = 0;
    v16bf awn[8];
    if (s < 4) { sel(wave * 5 + s + 1, Wn, biasn, o0n, moden); load_aw(Wn, o0n, awn); }

    v8f acc = {0.f,0.f,0.f,0.f,0.f,0.f,0.f,0.f};
    #pragma unroll
    for (int kc = 0; kc < 8; ++kc) acc = wmma_bf16(awc[kc], bx[kc], acc);

    if (mode == 2) {                                 // V -> [c][n] bf16
      #pragma unroll
      for (int g = 0; g < 8; ++g) {
        int o = o0 + g + 8 * half;                   // D: M = g + 8*(lane/16)
        vbf[(size_t)(b * CDIM + o) * NPIX + n0 + l16] = (bf16_t)(acc[g] + bias[o]);
      }
    } else {                                         // q/k -> [n][cr] bf16
      bf16_t* dst = (mode == 0) ? qT : kT;
      const float scale = (mode == 0) ? QSCALE : 1.0f;
      uint32_t* drow = (uint32_t*)(dst + (size_t)(b * NPIX + n0 + l16) * CRDIM);
      #pragma unroll
      for (int u = 0; u < 4; ++u) {
        int oA = o0 + 8 * half + 2 * u;
        drow[(o0 + 8 * half) / 2 + u] =
            pack2((acc[2*u]   + bias[oA])     * scale,
                  (acc[2*u+1] + bias[oA + 1]) * scale);
      }
    }

    if (s < 4) {
      W = Wn; bias = biasn; o0 = o0n; mode = moden;
      #pragma unroll
      for (int kc = 0; kc < 8; ++kc) awc[kc] = awn[kc];
    }
  }
}

// ---------------------------------------------------------------------------
// Flash-style attention, two-deep software pipeline:
//   scores(j) use K tiles prefetched last iteration; P.V(j-32) uses V tiles
//   prefetched last iteration; K(j+32)/V(j) loads issue before make_bp(j).
// Per 32-j chunk: 2 score WMMAs (S^T orientation), exp2(clip) -> bf16-pair
// pack -> 4 dword lane^16 swaps to form the P B-operand, 4 P.V WMMAs.
// Writes preT [(b*N+i)][c] bf16 (B-operand layout for the out-projection).
// ---------------------------------------------------------------------------
__global__ __launch_bounds__(128) void attn_kernel(
    const bf16_t* __restrict__ qT, const bf16_t* __restrict__ kT,
    const bf16_t* __restrict__ vbf, bf16_t* __restrict__ preT)
{
  const int tile = blockIdx.x;                      // BATCH * 256 i-tiles
  const int b    = tile >> 8;
  const int i0   = (tile & 255) << 4;
  const int lane = threadIdx.x & 31;
  const int wave = threadIdx.x >> 5;
  const int half = lane >> 4;
  const int l16  = lane & 15;

  const bf16_t* qp = qT + (size_t)(b * NPIX + i0 + l16) * CRDIM + (half ? 8 : 0);
  const v16bf bq = cat8(*(const v8bf*)qp, *(const v8bf*)(qp + 16));

  v8f acc[4];
  #pragma unroll
  for (int cc = 0; cc < 4; ++cc) acc[cc] = (v8f){0.f,0.f,0.f,0.f,0.f,0.f,0.f,0.f};
  float rs = 0.0f;

  const v8f zc = {0.f,0.f,0.f,0.f,0.f,0.f,0.f,0.f};
  const bf16_t* kbp  = kT + (size_t)b * NPIX * CRDIM + (size_t)l16 * CRDIM + (half ? 8 : 0);
  const bf16_t* vrow = vbf + (size_t)(b * CDIM + wave * 64 + l16) * NPIX + (half ? 16 : 0);

  auto load_k = [&](int j0, v16bf* ak) {
    const bf16_t* kp0 = kbp + (size_t)j0 * CRDIM;
    const bf16_t* kp1 = kp0 + 16 * CRDIM;
    ak[0] = cat8(*(const v8bf*)kp0, *(const v8bf*)(kp0 + 16));
    ak[1] = cat8(*(const v8bf*)kp1, *(const v8bf*)(kp1 + 16));
  };
  auto load_v = [&](int j0, v16bf* av) {
    #pragma unroll
    for (int cc = 0; cc < 4; ++cc) {
      const bf16_t* vp = vrow + (size_t)(cc * 16) * NPIX + j0;
      av[cc] = *(const v16bf*)vp;
    }
  };

  auto make_bp = [&](v8f t0, v8f t1) -> v16bf {
    uint32_t dlo[4], dhi[4];
    #pragma unroll
    for (int u = 0; u < 4; ++u) {
      float a0 = __builtin_amdgcn_exp2f(fminf(fmaxf(t0[2*u],   -CLIP2), CLIP2));
      float a1 = __builtin_amdgcn_exp2f(fminf(fmaxf(t0[2*u+1], -CLIP2), CLIP2));
      float b0 = __builtin_amdgcn_exp2f(fminf(fmaxf(t1[2*u],   -CLIP2), CLIP2));
      float b1 = __builtin_amdgcn_exp2f(fminf(fmaxf(t1[2*u+1], -CLIP2), CLIP2));
      rs += (a0 + a1) + (b0 + b1);
      dlo[u] = pack2(a0, a1);                       // B-operand dword: K=2u,2u+1
      dhi[u] = pack2(b0, b1);
    }
    // One lane^16 shuffle per dword: pre-select what the partner half needs.
    //  lane<16 gets partner dlo (j=8..15); lane>=16 gets partner dhi (j=16..23).
    v8u w;
    #pragma unroll
    for (int u = 0; u < 4; ++u) {
      uint32_t src = half ? dlo[u] : dhi[u];
      uint32_t xs  = __shfl_xor(src, 16, 32);
      w[u]     = half ? xs     : dlo[u];
      w[u + 4] = half ? dhi[u] : xs;
    }
    return __builtin_bit_cast(v16bf, w);
  };

  auto pv = [&](const v16bf* av, v16bf bp) {
    #pragma unroll
    for (int cc = 0; cc < 4; ++cc) acc[cc] = wmma_bf16(av[cc], bp, acc[cc]);
  };

  v16bf aknxt[2], av[4];
  {
    v16bf ak[2];
    load_k(0, ak);
    v8f t0 = wmma_bf16(ak[0], bq, zc);
    v8f t1 = wmma_bf16(ak[1], bq, zc);
    load_k(32, aknxt);
    load_v(0, av);
    v16bf bp = make_bp(t0, t1);

    for (int j0 = 32; j0 < NPIX; j0 += 32) {
      v8f s0 = wmma_bf16(aknxt[0], bq, zc);          // K(j0), prefetched
      v8f s1 = wmma_bf16(aknxt[1], bq, zc);
      pv(av, bp);                                    // chunk j0-32
      const int jn = (j0 + 32 < NPIX) ? j0 + 32 : 32;  // harmless dummy on last
      load_k(jn, aknxt);
      load_v(j0, av);
      bp = make_bp(s0, s1);
    }
    pv(av, bp);                                      // last chunk
  }

  const float rinv = 1.0f / (rs + __shfl_xor(rs, 16, 32));  // >= N*2^-CLIP2 > 0

  // Store preT [(b*N+i)][c]: D pairs (g=2u,2u+1) are adjacent c -> dword stores.
  uint32_t* prow = (uint32_t*)(preT + (size_t)(b * NPIX + i0 + l16) * CDIM);
  const int cloc = wave * 64 + 8 * half;
  #pragma unroll
  for (int cc = 0; cc < 4; ++cc) {
    #pragma unroll
    for (int u = 0; u < 4; ++u)
      prow[(cloc + cc * 16) / 2 + u] =
          pack2(acc[cc][2*u] * rinv, acc[cc][2*u+1] * rinv);
  }
}

// ---------------------------------------------------------------------------
// WMMA out-projection + gamma residual:
//   out = clip(gamma)*(ow @ pre + ob) + x
// A = ow rows (M=o, K=c), B = preT rows (K=c, N=n); o-tile loop software-
// pipelined like proj_kernel; f32 coalesced stores.
// ---------------------------------------------------------------------------
__global__ __launch_bounds__(128) void outproj_kernel(
    const bf16_t* __restrict__ preT, const bf16_t* __restrict__ wo,
    const float* __restrict__ ob, const float* __restrict__ gamma,
    const float* __restrict__ x, float* __restrict__ out)
{
  const int tile = blockIdx.x;                      // BATCH * 256
  const int b    = tile >> 8;
  const int n0   = (tile & 255) << 4;
  const int lane = threadIdx.x & 31;
  const int wave = threadIdx.x >> 5;
  const int half = lane >> 4;
  const int l16  = lane & 15;

  const float gcl = fminf(fmaxf(gamma[0], -1.0f), 1.0f);

  const bf16_t* prow = preT + (size_t)(b * NPIX + n0 + l16) * CDIM + (half ? 16 : 0);
  v16bf bx[8];
  #pragma unroll
  for (int kc = 0; kc < 8; ++kc) bx[kc] = *(const v16bf*)(prow + kc * 32);

  auto load_aw = [&](int o0, v16bf* aw) {
    const bf16_t* wr = wo + (size_t)(o0 + l16) * CDIM + (half ? 16 : 0);
    #pragma unroll
    for (int kc = 0; kc < 8; ++kc)
      aw[kc] = *(const v16bf*)(wr + kc * 32);
  };

  int o0 = wave * 64;
  v16bf awc[8];
  load_aw(o0, awc);

  #pragma unroll
  for (int s = 0; s < 4; ++s) {
    v16bf awn[8];
    if (s < 3) load_aw(wave * 64 + (s + 1) * 16, awn);

    v8f acc = {0.f,0.f,0.f,0.f,0.f,0.f,0.f,0.f};
    #pragma unroll
    for (int kc = 0; kc < 8; ++kc) acc = wmma_bf16(awc[kc], bx[kc], acc);

    #pragma unroll
    for (int g = 0; g < 8; ++g) {
      int o = o0 + g + 8 * half;
      size_t idx = (size_t)(b * CDIM + o) * NPIX + n0 + l16;
      out[idx] = gcl * (acc[g] + ob[o]) + x[idx];
    }

    if (s < 3) {
      o0 = wave * 64 + (s + 1) * 16;
      #pragma unroll
      for (int kc = 0; kc < 8; ++kc) awc[kc] = awn[kc];
    }
  }
}

// ---------------------------------------------------------------------------
extern "C" void kernel_launch(void* const* d_in, const int* in_sizes, int n_in,
                              void* d_out, int out_size, void* d_ws, size_t ws_size,
                              hipStream_t stream)
{
  (void)in_sizes; (void)n_in; (void)out_size; (void)ws_size;

  const float* x     = (const float*)d_in[0];
  const float* qw    = (const float*)d_in[1];
  const float* qb    = (const float*)d_in[2];
  const float* kw    = (const float*)d_in[3];
  const float* kb    = (const float*)d_in[4];
  const float* vw    = (const float*)d_in[5];
  const float* vb    = (const float*)d_in[6];
  const float* ow    = (const float*)d_in[7];
  const float* ob    = (const float*)d_in[8];
  const float* gamma = (const float*)d_in[9];
  float* out = (float*)d_out;

  auto align256 = [](size_t v) { return (v + 255) & ~(size_t)255; };
  char* ws = (char*)d_ws;
  size_t off = 0;
  bf16_t* wqb  = (bf16_t*)(ws + off); off = align256(off + (size_t)CRDIM * CDIM * 2);
  bf16_t* wkb  = (bf16_t*)(ws + off); off = align256(off + (size_t)CRDIM * CDIM * 2);
  bf16_t* wvb  = (bf16_t*)(ws + off); off = align256(off + (size_t)CDIM * CDIM * 2);
  bf16_t* wob  = (bf16_t*)(ws + off); off = align256(off + (size_t)CDIM * CDIM * 2);
  bf16_t* xT   = (bf16_t*)(ws + off); off = align256(off + (size_t)BATCH * NPIX * CDIM * 2);
  bf16_t* qT   = (bf16_t*)(ws + off); off = align256(off + (size_t)BATCH * NPIX * CRDIM * 2);
  bf16_t* kTb  = (bf16_t*)(ws + off); off = align256(off + (size_t)BATCH * NPIX * CRDIM * 2);
  bf16_t* vbf  = (bf16_t*)(ws + off); off = align256(off + (size_t)BATCH * CDIM * NPIX * 2);
  bf16_t* preT = (bf16_t*)(ws + off); off = align256(off + (size_t)BATCH * NPIX * CDIM * 2);
  // total ~26.3 MiB

  wconv_kernel<<<CDIM * CDIM / 256, 256, 0, stream>>>(qw, kw, vw, ow,
                                                      wqb, wkb, wvb, wob);
  xpose_kernel<<<BATCH * 4 * 64, 256, 0, stream>>>(x, xT);
  proj_kernel<<<BATCH * 256, 128, 0, stream>>>(xT, wqb, wkb, wvb, qb, kb, vb,
                                               qT, kTb, vbf);
  attn_kernel<<<BATCH * 256, 128, 0, stream>>>(qT, kTb, vbf, preT);
  outproj_kernel<<<BATCH * 256, 128, 0, stream>>>(preT, wob, ob, gamma, x, out);
}
